// NTXentLoss_64768106823834
// MI455X (gfx1250) — compile-verified, matrix operand order checked by reading, not run on hardware
//
#include <hip/hip_runtime.h>
#include <hip/hip_bf16.h>
#include <math.h>

#define BSZ 2048
#define DIM 512
#define INV_T 10.0f     // 1/TEMP
#define CSHIFT 10.0f    // logits <= 1/T, so exp(x/T - 10) never overflows
#define EPSN 1e-8f
#define LDB_PAD 520     // 512 + 8 halves row padding (bank spread for ds_load_b128)

typedef _Float16 v16h __attribute__((ext_vector_type(16)));
typedef _Float16 v8h  __attribute__((ext_vector_type(8)));
typedef float    v8f  __attribute__((ext_vector_type(8)));

// ---------------------------------------------------------------- utilities

__device__ __forceinline__ float wave_sum32(float v) {
#pragma unroll
  for (int m = 16; m >= 1; m >>= 1) v += __shfl_xor(v, m, 32);
  return v;
}

// WMMA 16-bit A/B operand fragment from a row-major f16 row:
// lane L: matrix row = L&15 ; K halves {kb..kb+7} and {kb+16..kb+23},
// kb = ktile*32 + ((L>>4) ? 8 : 0). Two 16B loads per lane.
__device__ __forceinline__ v16h load_frag_g(const _Float16* __restrict__ row_base, int kb) {
  v8h lo = *(const v8h*)(row_base + kb);
  v8h hi = *(const v8h*)(row_base + kb + 16);
  v16h r;
#pragma unroll
  for (int j = 0; j < 8; ++j) { r[j] = lo[j]; r[8 + j] = hi[j]; }
  return r;
}

// ------------------------------------------------- kernel 1: row-normalize -> f16
__global__ void ntxent_normalize_kernel(const float* __restrict__ z0,
                                        const float* __restrict__ z1,
                                        const float* __restrict__ z2,
                                        const float* __restrict__ z3,
                                        _Float16* __restrict__ mats) {
  const int gr   = blockIdx.x * 8 + (threadIdx.x >> 5);
  const int lane = threadIdx.x & 31;
  const int mat  = gr >> 11;
  const int row  = gr & (BSZ - 1);
  const float* src = (mat == 0 ? z0 : mat == 1 ? z1 : mat == 2 ? z2 : z3) + (size_t)row * DIM;
  _Float16*    dst = mats + (size_t)mat * BSZ * DIM + (size_t)row * DIM;

  float x[16];
  float ss = 0.0f;
#pragma unroll
  for (int j = 0; j < 16; ++j) {
    x[j] = src[j * 32 + lane];
    ss += x[j] * x[j];
  }
  ss = wave_sum32(ss);
  const float inv = 1.0f / fmaxf(sqrtf(ss), EPSN);
#pragma unroll
  for (int j = 0; j < 16; ++j)
    dst[j * 32 + lane] = (_Float16)(x[j] * inv);
}

// ------------------------------------------------- kernel 2: 9 diagonal dots (f32)
// diag[0..8]*BSZ : dOO dHH dPP  pOH pOP pHP  eON eHN ePN
__global__ void ntxent_diag_kernel(const _Float16* __restrict__ mats,
                                   float* __restrict__ diag) {
  const int row  = blockIdx.x * 8 + (threadIdx.x >> 5);
  const int lane = threadIdx.x & 31;
  const _Float16* O = mats + (size_t)row * DIM;
  const _Float16* H = O + (size_t)1 * BSZ * DIM;
  const _Float16* P = O + (size_t)2 * BSZ * DIM;
  const _Float16* N = O + (size_t)3 * BSZ * DIM;

  float a[9];
#pragma unroll
  for (int k = 0; k < 9; ++k) a[k] = 0.0f;

#pragma unroll
  for (int j = 0; j < 16; ++j) {
    const int idx = j * 32 + lane;
    const float fo = (float)O[idx], fh = (float)H[idx];
    const float fp = (float)P[idx], fn = (float)N[idx];
    a[0] += fo * fo; a[1] += fh * fh; a[2] += fp * fp;
    a[3] += fo * fh; a[4] += fo * fp; a[5] += fh * fp;
    a[6] += fo * fn; a[7] += fh * fn; a[8] += fp * fn;
  }
#pragma unroll
  for (int k = 0; k < 9; ++k) a[k] = wave_sum32(a[k]);
  if (lane == 0) {
#pragma unroll
    for (int k = 0; k < 9; ++k) diag[k * BSZ + row] = a[k];
  }
}

// ------------------------------------------------- kernel 3: fused gram + exp-reduce
// sums layout: 0 R_OO, 1 R_OH, 2 C_OH, 3 R_OP, 4 C_OP, 5 R_HH, 6 R_HP, 7 C_HP, 8 R_PP
// One block = (pair, 128-row block, 16-ct block). Each wave owns a 16-row A tile
// (fragments resident in VGPRs); the B strip is shared block-wide and pipelined
// through a 2x LDS buffer with GLOBAL_LOAD_ASYNC_TO_LDS_B128 (ASYNCcnt).
__global__ __launch_bounds__(256, 1)
void ntxent_gram_kernel(const _Float16* __restrict__ mats,
                        float* __restrict__ sums) {
  const int pair = blockIdx.x >> 7;
  const int rem  = blockIdx.x & 127;
  const int rowB = rem >> 3;          // 16 row-blocks of 128 rows
  const int ctB  = rem & 7;           // 8 ct-blocks of 16 column tiles
  const int wave = threadIdx.x >> 5;
  const int lane = threadIdx.x & 31;

  const int xIdx[6] = {0, 0, 0, 1, 1, 2};
  const int yIdx[6] = {0, 1, 2, 1, 2, 2};
  const int rIdx[6] = {0, 1, 3, 5, 6, 8};
  const int cIdx[6] = {-1, 2, 4, -1, 7, -1};

  const _Float16* X = mats + (size_t)xIdx[pair] * BSZ * DIM;
  const _Float16* Y = mats + (size_t)yIdx[pair] * BSZ * DIM;
  float* R  = sums + rIdx[pair] * BSZ;
  float* Cc = (cIdx[pair] >= 0) ? (sums + cIdx[pair] * BSZ) : nullptr;

  __shared__ _Float16 lB[2][16 * LDB_PAD];     // double-buffered B strip (2x16.6KB)

  const int mrow  = lane & 15;         // tile row (A) / tile col (B) for this lane
  const int khalf = (lane >> 4) << 3;  // K sub-chunk select per lane half
  const int row0  = rowB * 128 + wave * 16;

  // ---- per-wave A fragments: 16 kt tiles resident in VGPRs ----
  const _Float16* arow = X + (size_t)(row0 + mrow) * DIM;
  v16h afrag[16];
#pragma unroll
  for (int kt = 0; kt < 16; ++kt)
    afrag[kt] = load_frag_g(arow, kt * 32 + khalf);

  const int ct0 = ctB * 16;

  // cooperative async copy of one 16x512 B strip into LDS buffer `buf`
  auto issue_async_strip = [&](int buf, int ct) {
#pragma unroll
    for (int j = 0; j < 4; ++j) {                 // 1024 16B chunks / 256 threads
      const int c   = threadIdx.x + 256 * j;
      const int row = c >> 6;                     // 64 chunks per row
      const int k   = (c & 63) * 8;
      const unsigned long long ga =
          (unsigned long long)(size_t)(Y + (size_t)(ct * 16 + row) * DIM + k);
      const unsigned int ldsoff =
          (unsigned int)(size_t)&lB[buf][row * LDB_PAD + k];   // addr[31:0] = LDS addr
      asm volatile("global_load_async_to_lds_b128 %0, %1, off"
                   :: "v"(ldsoff), "v"(ga) : "memory");
    }
  };

  issue_async_strip(0, ct0);

  float rsum[8];
#pragma unroll
  for (int v = 0; v < 8; ++v) rsum[v] = 0.0f;

  for (int i = 0; i < 16; ++i) {
    const int ct = ct0 + i;
    asm volatile("s_wait_asynccnt 0x0" ::: "memory");  // our strip writes landed
    __syncthreads();                                   // everyone's landed
    if (i < 15) issue_async_strip((i + 1) & 1, ct + 1);

    const _Float16* bbase = &lB[i & 1][mrow * LDB_PAD];
    v8f acc = {};
#pragma unroll
    for (int kt = 0; kt < 16; ++kt) {
      const int kb = kt * 32 + khalf;
      v8h blo = *(const v8h*)(bbase + kb);
      v8h bhi = *(const v8h*)(bbase + kb + 16);
      v16h b;
#pragma unroll
      for (int j = 0; j < 8; ++j) { b[j] = blo[j]; b[8 + j] = bhi[j]; }
      acc = __builtin_amdgcn_wmma_f32_16x16x32_f16(false, afrag[kt], false, b,
                                                   (short)0, acc, false, false);
    }
    // acc[v] = G[m][n], m = v + 8*(lane>>4), n = lane&15 (cols ct*16 + n)
    float csum = 0.0f;
#pragma unroll
    for (int v = 0; v < 8; ++v) {
      const float e = __expf(acc[v] * INV_T - CSHIFT);
      rsum[v] += e;            // per-lane partial of rows (v, 8+v); reduce deferred
      csum += e;               // this lane's column partial over 8 rows
    }
    if (Cc) {
      csum += __shfl_xor(csum, 16, 32);                // combine the two row halves
      if (lane < 16) atomicAdd(Cc + ct * 16 + lane, csum);
    }
    // No trailing barrier needed: each wave's ds reads are consumed by its WMMAs
    // (s_wait_dscnt precedes the next top-of-loop barrier in program order), so
    // the back buffer is free when the next iteration's asyncs are issued.
  }

  // one cross-lane reduction after all 16 strips: sum over the 16 columns held by lanes
#pragma unroll
  for (int v = 0; v < 8; ++v) {
    float s = rsum[v];
#pragma unroll
    for (int m = 8; m >= 1; m >>= 1) s += __shfl_xor(s, m, 16);
    if ((lane & 15) == v)   // lane v -> row v ; lane 16+v -> row 8+v
      atomicAdd(R + row0 + v + ((lane >> 4) << 3), s);
  }
}

// ------------------------------------------------- kernel 4: assemble 3 losses
__global__ void ntxent_final_kernel(const float* __restrict__ sums,
                                    float* __restrict__ out) {
  const float* R_OO = sums + 0 * BSZ;
  const float* R_OH = sums + 1 * BSZ;
  const float* C_OH = sums + 2 * BSZ;
  const float* R_OP = sums + 3 * BSZ;
  const float* C_OP = sums + 4 * BSZ;
  const float* R_HH = sums + 5 * BSZ;
  const float* R_HP = sums + 6 * BSZ;
  const float* C_HP = sums + 7 * BSZ;
  const float* R_PP = sums + 8 * BSZ;
  const float* DG   = sums + 9 * BSZ;  // dOO dHH dPP pOH pOP pHP eON eHN ePN

  float l1 = 0.0f, l2 = 0.0f, l3 = 0.0f;
  for (int i = threadIdx.x; i < BSZ; i += blockDim.x) {
    const float Eoo = __expf(DG[0 * BSZ + i] * INV_T - CSHIFT);
    const float Ehh = __expf(DG[1 * BSZ + i] * INV_T - CSHIFT);
    const float Epp = __expf(DG[2 * BSZ + i] * INV_T - CSHIFT);
    const float poh = DG[3 * BSZ + i] * INV_T;
    const float pop = DG[4 * BSZ + i] * INV_T;
    const float php = DG[5 * BSZ + i] * INV_T;
    const float Eon = __expf(DG[6 * BSZ + i] * INV_T - CSHIFT);
    const float Ehn = __expf(DG[7 * BSZ + i] * INV_T - CSHIFT);
    const float Epn = __expf(DG[8 * BSZ + i] * INV_T - CSHIFT);

    // loss1: rows i and B+i (ori <-> hard_pos, extra col o.n)
    l1 += __logf(R_OO[i] + R_OH[i] - Eoo + Eon) + CSHIFT - poh;
    l1 += __logf(C_OH[i] + R_HH[i] - Ehh + Eon) + CSHIFT - poh;
    // loss2: ori <-> soft_pos, extra col o.n
    l2 += __logf(R_OO[i] + R_OP[i] - Eoo + Eon) + CSHIFT - pop;
    l2 += __logf(C_OP[i] + R_PP[i] - Epp + Eon) + CSHIFT - pop;
    // loss3: hard_pos <-> soft_pos, extras h.n and p.n
    l3 += __logf(R_HH[i] + R_HP[i] - Ehh + Ehn + Epn) + CSHIFT - php;
    l3 += __logf(C_HP[i] + R_PP[i] - Epp + Ehn + Epn) + CSHIFT - php;
  }

  __shared__ float s1[256], s2[256], s3[256];
  const int t = threadIdx.x;
  s1[t] = l1; s2[t] = l2; s3[t] = l3;
  __syncthreads();
  for (int off = 128; off >= 1; off >>= 1) {
    if (t < off) { s1[t] += s1[t + off]; s2[t] += s2[t + off]; s3[t] += s3[t + off]; }
    __syncthreads();
  }
  if (t == 0) {
    const float denom = 1.0f / (2.0f * (float)BSZ);
    out[0] = s1[0] * denom;
    out[1] = s2[0] * denom;
    out[2] = s3[0] * denom;
  }
}

// ---------------------------------------------------------------- launcher

extern "C" void kernel_launch(void* const* d_in, const int* in_sizes, int n_in,
                              void* d_out, int out_size, void* d_ws, size_t ws_size,
                              hipStream_t stream) {
  const float* z_ori = (const float*)d_in[0];
  const float* z_hp  = (const float*)d_in[1];
  const float* z_sp  = (const float*)d_in[2];
  const float* z_sn  = (const float*)d_in[3];

  _Float16* mats = (_Float16*)d_ws;                       // 4 * 2048 * 512 f16 = 8 MB
  float* sums = (float*)((char*)d_ws + (size_t)4 * BSZ * DIM * sizeof(_Float16));
  // sums: 9 reduction arrays (zeroed every call) + 9 diag arrays (fully overwritten)

  hipMemsetAsync(sums, 0, (size_t)9 * BSZ * sizeof(float), stream);

  ntxent_normalize_kernel<<<(4 * BSZ) / 8, 256, 0, stream>>>(z_ori, z_hp, z_sp, z_sn, mats);
  ntxent_diag_kernel<<<BSZ / 8, 256, 0, stream>>>(mats, sums + 9 * BSZ);
  // one block per (pair, 128-row block, 16-ct block): 6*16*8 = 768 blocks
  ntxent_gram_kernel<<<768, 256, 0, stream>>>(mats, sums);
  ntxent_final_kernel<<<1, 256, 0, stream>>>(sums, (float*)d_out);
}
